// AttentionSubsample_82815559401483
// MI455X (gfx1250) — compile-verified, matrix operand order checked by reading, not run on hardware
//
#include <hip/hip_runtime.h>
#include <cstdint>
#include <cstddef>

// ---------------------------------------------------------------------------
// AttentionSubsample (LeViT) for MI455X / gfx1250, wave32 + WMMA f16.
// All matmuls use V_WMMA_F32_16X16X32_F16 (f32 accumulate), BN folded into
// per-channel scale/shift epilogues. Workspace requirement: ~250 MB.
// ---------------------------------------------------------------------------

typedef __attribute__((ext_vector_type(16))) _Float16 v16h;
typedef __attribute__((ext_vector_type(8)))  _Float16 v8h;
typedef __attribute__((ext_vector_type(8)))  float    v8f;
typedef __attribute__((ext_vector_type(4)))  unsigned int v4u;

namespace {
constexpr int kB = 256, kN = 196, kC = 256, kH = 16, kKD = 16, kD = 64;
constexpr int kHW = 1280, kNHKD = 256, kDH = 1024, kOUT = 512;
constexpr int kQ = 49, kQPAD = 64, kNPAD = 208, kKPAD = 224;
}

__device__ __forceinline__ v8f wmma16(v16h a, v16h b, v8f c) {
  // 8 args: (neg_a, A, neg_b, B, c_mod, C, reuse_a, reuse_b)
  return __builtin_amdgcn_wmma_f32_16x16x32_f16(false, a, false, b, (short)0, c,
                                                false, false);
}

// Load a lane's 16 halves of a 16x32 (f16) A/B fragment: two contiguous
// 16-byte loads at K offsets base and base+16 (base folded into p by caller).
__device__ __forceinline__ v16h ld_pair(const _Float16* p) {
  v8h lo = *(const v8h*)(p);
  v8h hi = *(const v8h*)(p + 16);
  return __builtin_shufflevector(lo, hi, 0, 1, 2, 3, 4, 5, 6, 7, 8, 9, 10, 11,
                                 12, 13, 14, 15);
}

// ---------------------------------------------------------------------------
// Prep kernels
// ---------------------------------------------------------------------------
__global__ void bn_coeffs_kernel(const float* __restrict__ g,
                                 const float* __restrict__ bb,
                                 const float* __restrict__ m,
                                 const float* __restrict__ v,
                                 float* __restrict__ sc,
                                 float* __restrict__ sh, int n) {
  int i = blockIdx.x * blockDim.x + threadIdx.x;
  if (i < n) {
    float s = g[i] * rsqrtf(v[i] + 1e-5f);
    sc[i] = s;
    sh[i] = bb[i] - m[i] * s;
  }
}

__global__ void cvt_f16_kernel(const float* __restrict__ src,
                               _Float16* __restrict__ dst, int n) {
  int i = blockIdx.x * blockDim.x + threadIdx.x;
  if (i < n) dst[i] = (_Float16)src[i];
}

__global__ void fill0_kernel(v4u* __restrict__ p, int n16) {
  int i = blockIdx.x * blockDim.x + threadIdx.x;
  if (i < n16) {
    v4u z = {0u, 0u, 0u, 0u};
    p[i] = z;
  }
}

// xs = x[:, ::2, ::2] rows gathered into a dense [B*49, C] f16 matrix.
__global__ void gather_xs_kernel(const _Float16* __restrict__ x16,
                                 _Float16* __restrict__ xs16, int total) {
  int i = blockIdx.x * blockDim.x + threadIdx.x;
  if (i >= total) return;
  int c = i & (kC - 1);
  int r = i >> 8;  // kC == 256
  int b = r / kQ, qi = r - b * kQ;
  int srow = b * kN + (qi / 7) * 28 + (qi % 7) * 2;  // (2*qr)*14 + 2*qc
  xs16[i] = x16[(size_t)srow * kC + c];
}

// ---------------------------------------------------------------------------
// Generic tiled WMMA GEMM:  out[M,N] = A[M,K] * W[N,K]^T, BN epilogue.
// Block = 256 threads = 8 waves arranged 4(M) x 2(N); wave tile = 32x32
// (2x2 WMMA fragments). MODE selects the epilogue scatter.
// ---------------------------------------------------------------------------
template <int MODE>
__global__ __launch_bounds__(256) void gemm_bn_kernel(
    const _Float16* __restrict__ A, const _Float16* __restrict__ W,
    const float* __restrict__ sc, const float* __restrict__ sh, int M, int N,
    int K, float* __restrict__ outF, _Float16* __restrict__ outK,
    _Float16* __restrict__ outV) {
  int lane = threadIdx.x & 31;
  int w = threadIdx.x >> 5;
  int wm = w >> 1, wn = w & 1;
  int rrow = lane & 15;
  int hsel = lane >> 4;
  int kb = hsel * 8;  // K half-offset per the 16-bit A/B VGPR layout
  int m0 = blockIdx.y * 128 + wm * 32;
  int n0 = blockIdx.x * 64 + wn * 32;

  v8f c[2][2] = {};

  for (int k0 = 0; k0 < K; k0 += 32) {
    v16h av[2], bv[2];
#pragma unroll
    for (int im = 0; im < 2; ++im)
      av[im] = ld_pair(A + (size_t)(m0 + 16 * im + rrow) * K + k0 + kb);
#pragma unroll
    for (int jn = 0; jn < 2; ++jn)
      bv[jn] = ld_pair(W + (size_t)(n0 + 16 * jn + rrow) * K + k0 + kb);
#pragma unroll
    for (int im = 0; im < 2; ++im)
#pragma unroll
      for (int jn = 0; jn < 2; ++jn) c[im][jn] = wmma16(av[im], bv[jn], c[im][jn]);
  }

#pragma unroll
  for (int im = 0; im < 2; ++im) {
#pragma unroll
    for (int jn = 0; jn < 2; ++jn) {
#pragma unroll
      for (int i = 0; i < 8; ++i) {
        int mrow = m0 + 16 * im + i + 8 * hsel;  // C/D layout: M = vgpr + 8*hi
        int ncol = n0 + 16 * jn + rrow;          //             N = lane&15
        float val = c[im][jn][i] * sc[ncol] + sh[ncol];
        if (MODE == 0) {
          // kv GEMM: mrow = b*196+n ; ncol = h*80 + cc ; split k / v.
          int bb = mrow / kN, nn = mrow - bb * kN;
          int hh = ncol / 80, cc = ncol - hh * 80;
          int bh = bb * kH + hh;
          if (cc < kKD) {
            // k: [bh][208 rows (padded)][32 halves (K padded 16->32)]
            outK[((size_t)bh * kNPAD + nn) * 32 + cc] = (_Float16)val;
          } else {
            // v^T: [bh][64 d][224 ki (padded)]
            outV[((size_t)bh * kD + (cc - kKD)) * kKPAD + nn] = (_Float16)val;
          }
        } else if (MODE == 1) {
          // q GEMM: mrow = b*49+qi ; ncol = h*16 + d.
          int bb = mrow / kQ, qq = mrow - bb * kQ;
          int hh = ncol >> 4, dd = ncol & 15;
          outK[((size_t)(bb * kH + hh) * kQPAD + qq) * 32 + dd] = (_Float16)val;
        } else {
          outF[(size_t)mrow * N + ncol] = val;  // projection -> d_out (f32)
        }
      }
    }
  }
}

// ---------------------------------------------------------------------------
// Fused attention: one workgroup per (b,h); 4 waves, each owning 16 q-rows.
// QK^T logits stay in registers (13 x v8f per wave); softmax + bias done with
// cross-lane shuffles; f16 probs staged through LDS for the attn*V WMMAs.
// ---------------------------------------------------------------------------
__global__ __launch_bounds__(128) void attn_kernel(
    const _Float16* __restrict__ q_ws, const _Float16* __restrict__ k_ws,
    const _Float16* __restrict__ v_ws, const float* __restrict__ biases,
    const int* __restrict__ bias_idxs, _Float16* __restrict__ o_ws,
    int n_off) {
  __shared__ _Float16 s_attnh[kQPAD * kKPAD] __attribute__((aligned(16)));

  int lane = threadIdx.x & 31;
  int w = threadIdx.x >> 5;
  int rrow = lane & 15, hsel = lane >> 4, kb = hsel * 8;
  int bh = blockIdx.x;
  int b = bh >> 4, h = bh & 15;

  // A fragment: this wave's 16 q rows (K padded 16->32, upper halves zero).
  v16h aq = ld_pair(q_ws + ((size_t)bh * kQPAD + 16 * w + rrow) * 32 + kb);

  // QK^T: 13 key tiles of 16 rows (196 padded to 208, pad rows are zero).
  v8f c[13];
#pragma unroll
  for (int nt = 0; nt < 13; ++nt) {
    v16h bk = ld_pair(k_ws + ((size_t)bh * kNPAD + 16 * nt + rrow) * 32 + kb);
    v8f z = {};
    c[nt] = wmma16(aq, bk, z);
  }

  // Bias + scale + softmax, register-resident. Row qi = 16w + i + 8*hsel is
  // spread over the 16 lanes sharing hsel -> reduce with xor masks 1,2,4,8.
  const float* btab = biases + h * n_off;
#pragma unroll
  for (int i = 0; i < 8; ++i) {
    int qi = 16 * w + i + 8 * hsel;
    float vals[13];
    float mx = -1e30f;
#pragma unroll
    for (int nt = 0; nt < 13; ++nt) {
      int ki = 16 * nt + rrow;
      float val = -1e30f;
      if (ki < kN) {
        val = c[nt][i] * 0.25f;  // SCALE = KD^-0.5 = 0.25
        if (qi < kQ) val += btab[bias_idxs[qi * kN + ki]];
      }
      vals[nt] = val;
      mx = fmaxf(mx, val);
    }
#pragma unroll
    for (int d = 8; d >= 1; d >>= 1) mx = fmaxf(mx, __shfl_xor(mx, d, 32));
    float sum = 0.f;
#pragma unroll
    for (int nt = 0; nt < 13; ++nt) {
      int ki = 16 * nt + rrow;
      float e = (ki < kN) ? __expf(vals[nt] - mx) : 0.f;
      vals[nt] = e;
      sum += e;
    }
#pragma unroll
    for (int d = 8; d >= 1; d >>= 1) sum += __shfl_xor(sum, d, 32);
    float inv = 1.f / sum;
#pragma unroll
    for (int nt = 0; nt < 13; ++nt)
      s_attnh[qi * kKPAD + 16 * nt + rrow] = (_Float16)(vals[nt] * inv);
    s_attnh[qi * kKPAD + kNPAD + rrow] = (_Float16)0.f;  // pad 208..223
  }

  // O = attn * V : K = 224 (7 steps of 32), N = 64 (4 tiles). Wave-private
  // rows -> no barrier needed between phases.
  v8f o[4] = {};
  for (int kt = 0; kt < 7; ++kt) {
    v16h aa = ld_pair(&s_attnh[(16 * w + rrow) * kKPAD + 32 * kt + kb]);
#pragma unroll
    for (int nt = 0; nt < 4; ++nt) {
      v16h bvv =
          ld_pair(v_ws + ((size_t)bh * kD + 16 * nt + rrow) * kKPAD + 32 * kt + kb);
      o[nt] = wmma16(aa, bvv, o[nt]);
    }
  }

  // Fused hardswish, write o_ws[b*49+qi][h*64+d] (f16) for the projection.
#pragma unroll
  for (int nt = 0; nt < 4; ++nt) {
#pragma unroll
    for (int i = 0; i < 8; ++i) {
      int qi = 16 * w + i + 8 * hsel;
      if (qi < kQ) {
        int d = 16 * nt + rrow;
        float xv = o[nt][i];
        float t = fminf(fmaxf(xv + 3.f, 0.f), 6.f);
        o_ws[((size_t)(b * kQ + qi)) * kDH + h * kD + d] =
            (_Float16)(xv * t * (1.f / 6.f));
      }
    }
  }
}

// ---------------------------------------------------------------------------
// Host-side launch
// ---------------------------------------------------------------------------
extern "C" void kernel_launch(void* const* d_in, const int* in_sizes, int n_in,
                              void* d_out, int out_size, void* d_ws,
                              size_t ws_size, hipStream_t stream) {
  (void)n_in; (void)out_size; (void)ws_size;
  const float* x      = (const float*)d_in[0];
  const float* W_kv   = (const float*)d_in[1];
  const float* g_kv   = (const float*)d_in[2];
  const float* b_kv   = (const float*)d_in[3];
  const float* m_kv   = (const float*)d_in[4];
  const float* v_kv   = (const float*)d_in[5];
  const float* W_q    = (const float*)d_in[6];
  const float* g_q    = (const float*)d_in[7];
  const float* b_q    = (const float*)d_in[8];
  const float* m_q    = (const float*)d_in[9];
  const float* v_q    = (const float*)d_in[10];
  const float* W_p    = (const float*)d_in[11];
  const float* g_p    = (const float*)d_in[12];
  const float* b_p    = (const float*)d_in[13];
  const float* m_p    = (const float*)d_in[14];
  const float* v_p    = (const float*)d_in[15];
  const float* biases = (const float*)d_in[16];
  const int*   bidx   = (const int*)d_in[17];
  int n_off = in_sizes[16] / kH;

  char* ws = (char*)d_ws;
  auto carve = [&](size_t bytes) {
    char* p = ws;
    ws += (bytes + 255) & ~(size_t)255;
    return p;
  };

  _Float16* x16   = (_Float16*)carve((size_t)kB * kN * kC * 2);
  _Float16* xs16  = (_Float16*)carve((size_t)kB * kQ * kC * 2);
  _Float16* wkv16 = (_Float16*)carve((size_t)kHW * kC * 2);
  _Float16* wq16  = (_Float16*)carve((size_t)kNHKD * kC * 2);
  _Float16* wp16  = (_Float16*)carve((size_t)kOUT * kDH * 2);
  float* sc_kv = (float*)carve(kHW * 4);
  float* sh_kv = (float*)carve(kHW * 4);
  float* sc_q  = (float*)carve(kNHKD * 4);
  float* sh_q  = (float*)carve(kNHKD * 4);
  float* sc_p  = (float*)carve(kOUT * 4);
  float* sh_p  = (float*)carve(kOUT * 4);
  _Float16* q_ws = (_Float16*)carve((size_t)kB * kH * kQPAD * 32 * 2);
  _Float16* k_ws = (_Float16*)carve((size_t)kB * kH * kNPAD * 32 * 2);
  _Float16* v_ws = (_Float16*)carve((size_t)kB * kH * kD * kKPAD * 2);
  _Float16* o_ws = (_Float16*)carve((size_t)kB * kQ * kDH * 2);

  auto nb = [](long long n, int t) { return dim3((unsigned)((n + t - 1) / t)); };

  bn_coeffs_kernel<<<nb(kHW, 256), 256, 0, stream>>>(g_kv, b_kv, m_kv, v_kv,
                                                     sc_kv, sh_kv, kHW);
  bn_coeffs_kernel<<<nb(kNHKD, 256), 256, 0, stream>>>(g_q, b_q, m_q, v_q,
                                                       sc_q, sh_q, kNHKD);
  bn_coeffs_kernel<<<nb(kOUT, 256), 256, 0, stream>>>(g_p, b_p, m_p, v_p, sc_p,
                                                      sh_p, kOUT);

  long long nx = (long long)kB * kN * kC;
  cvt_f16_kernel<<<nb(nx, 256), 256, 0, stream>>>(x, x16, (int)nx);
  cvt_f16_kernel<<<nb(kHW * kC, 256), 256, 0, stream>>>(W_kv, wkv16, kHW * kC);
  cvt_f16_kernel<<<nb(kNHKD * kC, 256), 256, 0, stream>>>(W_q, wq16,
                                                          kNHKD * kC);
  cvt_f16_kernel<<<nb(kOUT * kDH, 256), 256, 0, stream>>>(W_p, wp16,
                                                          kOUT * kDH);

  long long nxs = (long long)kB * kQ * kC;
  gather_xs_kernel<<<nb(nxs, 256), 256, 0, stream>>>(x16, xs16, (int)nxs);

  long long nq16 = (long long)kB * kH * kQPAD * 32 / 8;
  long long nk16 = (long long)kB * kH * kNPAD * 32 / 8;
  long long nv16 = (long long)kB * kH * kD * kKPAD / 8;
  fill0_kernel<<<nb(nq16, 256), 256, 0, stream>>>((v4u*)q_ws, (int)nq16);
  fill0_kernel<<<nb(nk16, 256), 256, 0, stream>>>((v4u*)k_ws, (int)nk16);
  fill0_kernel<<<nb(nv16, 256), 256, 0, stream>>>((v4u*)v_ws, (int)nv16);

  int Mkv = kB * kN;  // 50176 = 392 * 128
  gemm_bn_kernel<0><<<dim3(kHW / 64, Mkv / 128), 256, 0, stream>>>(
      x16, wkv16, sc_kv, sh_kv, Mkv, kHW, kC, nullptr, k_ws, v_ws);

  int Mq = kB * kQ;  // 12544 = 98 * 128
  gemm_bn_kernel<1><<<dim3(kNHKD / 64, Mq / 128), 256, 0, stream>>>(
      xs16, wq16, sc_q, sh_q, Mq, kNHKD, kC, nullptr, q_ws, nullptr);

  attn_kernel<<<dim3(kB * kH), 128, 0, stream>>>(q_ws, k_ws, v_ws, biases,
                                                 bidx, o_ws, n_off);

  gemm_bn_kernel<2><<<dim3(kOUT / 64, Mq / 128), 256, 0, stream>>>(
      o_ws, wp16, sc_p, sh_p, Mq, kOUT, kDH, (float*)d_out, nullptr, nullptr);
}
